// PDENET_68805376082716
// MI455X (gfx1250) — matrix operand order repaired and checked by reading, not compiled, not run on hardware
//
#include <hip/hip_runtime.h>
#include <hip/hip_bf16.h>
#include <stdint.h>
#include <stddef.h>

// ---------------------------------------------------------------------------
// Schrödinger-PINN loss, fully fused per 64-row tile on CDNA5 (gfx1250).
//   streams: 0 = h (primal), 1 = dh/dx, 2 = dh/dt, 3 = d2h/dx2
//   hidden layers: v_wmma_f32_16x16x32_f16, activations resident in LDS,
//   weight tiles staged with global_load_async_to_lds_b128 (ASYNCcnt).
// ---------------------------------------------------------------------------

#define H    200
#define KP   224      // padded hidden dim: 14 x 16 (N) == 7 x 32 (K)
#define ROWS 64       // rows per workgroup (4 M-tiles x 16 rows)
#define NT   14       // N-tiles (split even/odd across wave pairs)
#define KTL  7        // K-tiles

#define N0_CNT 4096
#define NB_CNT 4096
#define NF_CNT 131072

typedef _Float16 v16h __attribute__((ext_vector_type(16)));
typedef _Float16 v8h  __attribute__((ext_vector_type(8)));
typedef float    v8f  __attribute__((ext_vector_type(8)));
typedef __attribute__((address_space(3))) char lds_char;

// Build a 16-wide f16 fragment from two 8-element K runs (two ds_load_b128),
// matching the CDNA5 16-bit A/B fragment lane layouts.
__device__ __forceinline__ v16h load_frag16(const _Float16* base) {
  v8h lo = *(const v8h*)(base);
  v8h hi = *(const v8h*)(base + 16);
  return __builtin_shufflevector(lo, hi, 0,1,2,3,4,5,6,7,8,9,10,11,12,13,14,15);
}

// Native gfx1250 v_tanh_f32 when the builtin exists; otherwise a branch-free
// exp-based tanh (no EXEC divergence inside the WMMA region).
__device__ __forceinline__ float fast_tanh(float x) {
#if __has_builtin(__builtin_amdgcn_tanhf)
  return __builtin_amdgcn_tanhf(x);
#elif __has_builtin(__builtin_amdgcn_tanh_f32)
  return __builtin_amdgcn_tanh_f32(x);
#else
  float ax = fabsf(x);
  float e  = __expf(-2.0f * ax);
  float r  = (1.0f - e) * __builtin_amdgcn_rcpf(1.0f + e);
  return copysignf(r, x);
#endif
}

// Async global -> LDS copy of one 16-byte chunk (per-lane), ISA operand
// order: VDST = LDS byte address, ADDR = 64-bit global address.
__device__ __forceinline__ void async_copy_b128(uint32_t lds_byte_off,
                                                const void* gptr) {
  asm volatile("global_load_async_to_lds_b128 %0, %1, off"
               :: "v"(lds_byte_off), "v"((uint64_t)(uintptr_t)gptr)
               : "memory");
}

__device__ __forceinline__ void wait_asynccnt0() {
#if __has_builtin(__builtin_amdgcn_s_wait_asynccnt)
  __builtin_amdgcn_s_wait_asynccnt(0);
#else
  asm volatile("s_wait_asynccnt 0x0" ::: "memory");
#endif
}

// ---------------------------------------------------------------------------
// Weight prep: transpose W2/W3/W4 ([k][n] f32) into Wt[L][n][k] f16 padded
// to KP x KP. Padded rows/cols are zero, so the extra N-tile writes exact
// zeros through the tanh chains (tanh(0)=0, s*0=0) -> no explicit pad init.
// ---------------------------------------------------------------------------
__global__ __launch_bounds__(256)
void prep_weights(const float* __restrict__ W2, const float* __restrict__ W3,
                  const float* __restrict__ W4, _Float16* __restrict__ Wt) {
  int idx = blockIdx.x * blockDim.x + threadIdx.x;
  const int per = KP * KP;
  if (idx >= 3 * per) return;
  int L = idx / per, rem = idx % per;
  int n = rem / KP, k = rem % KP;
  const float* W = (L == 0) ? W2 : (L == 1) ? W3 : W4;
  float v = (n < H && k < H) ? W[k * H + n] : 0.0f;
  Wt[idx] = (_Float16)v;
}

// ---------------------------------------------------------------------------
// Fully fused MLP + forward-mode JVP streams. 256 threads = 8 waves:
//   wave>>1 selects the 16-row M-tile, wave&1 selects even/odd N-tiles.
// Dynamic LDS:
//   act   : 2 (ping-pong) * NS * ROWS * KP   f16
//   btile : 2 * 16 * KP                      f16  (even/odd N-tile pair)
//   bias  : KP                               f32
// ---------------------------------------------------------------------------
template <int NS>
__global__ __launch_bounds__(256)
void fused_mlp(const float* __restrict__ X,
               const _Float16* __restrict__ Wt,
               const float* __restrict__ W1, const float* __restrict__ b1,
               const float* __restrict__ b2, const float* __restrict__ b3,
               const float* __restrict__ b4,
               const float* __restrict__ W5, const float* __restrict__ b5,
               float* __restrict__ out) {
  extern __shared__ char smem_raw[];
  _Float16* act   = (_Float16*)smem_raw;                       // 2*NS*ROWS*KP
  _Float16* btile = act + (size_t)2 * NS * ROWS * KP;          // 2*16*KP
  float*    biasL = (float*)(btile + 2 * 16 * KP);             // KP

  const int tid     = threadIdx.x;
  const int lane    = tid & 31;
  const int wave    = tid >> 5;
  const int rowBase = (wave >> 1) * 16;
  const int ntHalf  = wave & 1;
  const int nLane   = lane & 15;
  const bool hiHalf = lane >= 16;

  _Float16* bufA = act;
  _Float16* bufB = act + (size_t)NS * ROWS * KP;
  const uint32_t btileOff = (uint32_t)(uintptr_t)(lds_char*)btile;

  // ---- layer 1 (2 -> 200): tiny K, plain VALU; writes all KP columns -----
  const float* Xr = X + (size_t)blockIdx.x * ROWS * 2;
  for (int idx = tid; idx < ROWS * KP; idx += 256) {
    int r = idx / KP, n = idx % KP;
    float x  = Xr[r * 2 + 0];
    float tt = Xr[r * 2 + 1];
    float w0 = (n < H) ? W1[n]     : 0.0f;   // dz/dx
    float w1 = (n < H) ? W1[H + n] : 0.0f;   // dz/dt
    float bb = (n < H) ? b1[n]     : 0.0f;
    float z  = fmaf(x, w0, fmaf(tt, w1, bb));
    float a  = fast_tanh(z);
    float se = 1.0f - a * a;
    bufA[((size_t)0 * ROWS + r) * KP + n] = (_Float16)a;
    if (NS >= 2) bufA[((size_t)1 * ROWS + r) * KP + n] = (_Float16)(se * w0);
    if (NS >= 4) {
      bufA[((size_t)2 * ROWS + r) * KP + n] = (_Float16)(se * w1);
      bufA[((size_t)3 * ROWS + r) * KP + n] = (_Float16)(-2.0f * a * se * w0 * w0);
    }
  }

  // ---- hidden layers 2..4 (200 -> 200): WMMA ------------------------------
  for (int L = 0; L < 3; ++L) {
    _Float16* inb  = (L & 1) ? bufB : bufA;
    _Float16* outb = (L & 1) ? bufA : bufB;
    const _Float16* wl = Wt + (size_t)L * KP * KP;
    const float* bsrc = (L == 0) ? b2 : (L == 1) ? b3 : b4;

    __syncthreads();
    for (int i = tid; i < KP; i += 256) biasL[i] = (i < H) ? bsrc[i] : 0.0f;

    for (int p = 0; p < NT / 2; ++p) {
      __syncthreads();
      // async-stage the N-tile pair (32 x KP f16, contiguous) into LDS
      {
        const char* src = (const char*)(wl + (size_t)(2 * p) * 16 * KP);
        for (int i = tid; i < (2 * 16 * KP * 2) / 16; i += 256)
          async_copy_b128(btileOff + i * 16, src + (size_t)i * 16);
        if (p + 1 < NT / 2)  // pull next pair toward L2
          __builtin_prefetch(wl + (size_t)(2 * p + 2) * 16 * KP, 0, 1);
        wait_asynccnt0();
      }
      __syncthreads();

      const int nt = 2 * p + ntHalf;

      // B fragments for this wave's N-tile (reused across all NS streams)
      v16h bf[KTL];
      {
        const _Float16* brow = btile + ((size_t)ntHalf * 16 + nLane) * KP
                                     + (hiHalf ? 8 : 0);
        #pragma unroll
        for (int kt = 0; kt < KTL; ++kt) bf[kt] = load_frag16(brow + kt * 32);
      }

      float aT[8], zxT[8];
      #pragma unroll
      for (int s = 0; s < NS; ++s) {
        v8f acc = {};
        const _Float16* arow =
            inb + ((size_t)s * ROWS + rowBase + nLane) * KP + (hiHalf ? 8 : 0);
        #pragma unroll
        for (int kt = 0; kt < KTL; ++kt) {
          v16h af = load_frag16(arow + kt * 32);
          acc = __builtin_amdgcn_wmma_f32_16x16x32_f16(
              false, af, false, bf[kt], (short)0, acc, false, false);
        }
        const int col = nt * 16 + nLane;
        #pragma unroll
        for (int j = 0; j < 8; ++j) {
          int m = j + (hiHalf ? 8 : 0);
          _Float16* op = outb + ((size_t)s * ROWS + rowBase + m) * KP + col;
          float v = acc[j];
          if (s == 0) {                       // a = tanh(z + b)
            float a = fast_tanh(v + biasL[col]);
            aT[j] = a; *op = (_Float16)a;
          } else if (s == 1) {                // a_x = s * z_x  (keep z_x)
            float se = 1.0f - aT[j] * aT[j];
            zxT[j] = v; *op = (_Float16)(se * v);
          } else if (s == 2) {                // a_t = s * z_t
            float se = 1.0f - aT[j] * aT[j];
            *op = (_Float16)(se * v);
          } else {                            // a_xx = s*z_xx - 2 a s z_x^2
            float a = aT[j], se = 1.0f - a * a;
            *op = (_Float16)(se * v - 2.0f * a * se * zxT[j] * zxT[j]);
          }
        }
      }
    }
  }

  // ---- layer 5 (200 -> 2): tiny N, VALU reduction -------------------------
  __syncthreads();
  const _Float16* fin = bufB;  // layer L=2 wrote bufB
  if (tid < 128) {
    const int r = tid >> 1, o = tid & 1;
    #pragma unroll
    for (int s = 0; s < NS; ++s) {
      float accv = (s == 0) ? b5[o] : 0.0f;
      const _Float16* hp = fin + ((size_t)s * ROWS + r) * KP;
      for (int k = 0; k < H; ++k) accv = fmaf((float)hp[k], W5[k * 2 + o], accv);
      out[((size_t)(blockIdx.x * ROWS + r)) * (2 * NS) + s * 2 + o] = accv;
    }
  }
}

// ---------------------------------------------------------------------------
// Deterministic single-block finalize: 8 MSE terms -> scalar loss.
// out0: [N0][2]; outlb/outub: [Nb][4] (u,v,ux,vx);
// outf: [Nf][8] (u,v,ux,vx,ut,vt,uxx,vxx)
// ---------------------------------------------------------------------------
__global__ __launch_bounds__(256)
void finalize_loss(const float* __restrict__ out0,
                   const float* __restrict__ outlb,
                   const float* __restrict__ outub,
                   const float* __restrict__ outf,
                   const float* __restrict__ u0t,
                   const float* __restrict__ v0t,
                   float* __restrict__ loss) {
  __shared__ float red[8][256];
  const int t = threadIdx.x;
  float acc[8];
  #pragma unroll
  for (int q = 0; q < 8; ++q) acc[q] = 0.0f;

  for (int i = t; i < N0_CNT; i += 256) {
    float du = u0t[i] - out0[2 * i + 0];
    float dv = v0t[i] - out0[2 * i + 1];
    acc[0] = fmaf(du, du, acc[0]);
    acc[1] = fmaf(dv, dv, acc[1]);
  }
  for (int i = t; i < NB_CNT; i += 256) {
    const float* lb = outlb + 4 * i;
    const float* ub = outub + 4 * i;
    float d0 = lb[0] - ub[0], d1 = lb[1] - ub[1];
    float d2 = lb[2] - ub[2], d3 = lb[3] - ub[3];
    acc[2] = fmaf(d0, d0, acc[2]);
    acc[3] = fmaf(d1, d1, acc[3]);
    acc[4] = fmaf(d2, d2, acc[4]);
    acc[5] = fmaf(d3, d3, acc[5]);
  }
  for (int i = t; i < NF_CNT; i += 256) {
    const float* rw = outf + 8 * i;
    float u = rw[0], v = rw[1];
    float ut = rw[4], vt = rw[5], uxx = rw[6], vxx = rw[7];
    float sq = u * u + v * v;
    float fu = -vt + 0.5f * uxx + sq * u;
    float fv =  ut + 0.5f * vxx + sq * v;
    acc[6] = fmaf(fu, fu, acc[6]);
    acc[7] = fmaf(fv, fv, acc[7]);
  }

  #pragma unroll
  for (int q = 0; q < 8; ++q) red[q][t] = acc[q];
  __syncthreads();
  for (int off = 128; off > 0; off >>= 1) {
    if (t < off) {
      #pragma unroll
      for (int q = 0; q < 8; ++q) red[q][t] += red[q][t + off];
    }
    __syncthreads();
  }
  if (t == 0) {
    loss[0] = (red[0][0] + red[1][0]) * (1.0f / N0_CNT)
            + (red[2][0] + red[3][0] + red[4][0] + red[5][0]) * (1.0f / NB_CNT)
            + (red[6][0] + red[7][0]) * (1.0f / NF_CNT);
  }
}

// ---------------------------------------------------------------------------
// Host launcher
// ---------------------------------------------------------------------------
static inline size_t smem_bytes(int ns) {
  return (size_t)2 * ns * ROWS * KP * sizeof(_Float16)  // ping-pong acts
       + (size_t)2 * 16 * KP * sizeof(_Float16)         // B-tile pair
       + (size_t)KP * sizeof(float);                    // bias
}

extern "C" void kernel_launch(void* const* d_in, const int* in_sizes, int n_in,
                              void* d_out, int out_size, void* d_ws, size_t ws_size,
                              hipStream_t stream) {
  (void)in_sizes; (void)n_in; (void)out_size; (void)ws_size;

  const float* X0   = (const float*)d_in[0];
  const float* X_lb = (const float*)d_in[1];
  const float* X_ub = (const float*)d_in[2];
  const float* X_f  = (const float*)d_in[3];
  const float* u0t  = (const float*)d_in[4];
  const float* v0t  = (const float*)d_in[5];
  const float* W1 = (const float*)d_in[6];  const float* b1 = (const float*)d_in[7];
  const float* W2 = (const float*)d_in[8];  const float* b2 = (const float*)d_in[9];
  const float* W3 = (const float*)d_in[10]; const float* b3 = (const float*)d_in[11];
  const float* W4 = (const float*)d_in[12]; const float* b4 = (const float*)d_in[13];
  const float* W5 = (const float*)d_in[14]; const float* b5 = (const float*)d_in[15];

  const size_t OFF_WT    = 0;                                   // 3*KP*KP f16
  const size_t OFF_OUT0  = OFF_WT    + (size_t)3 * KP * KP * 2;
  const size_t OFF_OUTLB = OFF_OUT0  + (size_t)N0_CNT * 2 * 4;
  const size_t OFF_OUTUB = OFF_OUTLB + (size_t)NB_CNT * 4 * 4;
  const size_t OFF_OUTF  = OFF_OUTUB + (size_t)NB_CNT * 4 * 4;

  char* ws = (char*)d_ws;
  _Float16* wt = (_Float16*)(ws + OFF_WT);
  float* out0  = (float*)(ws + OFF_OUT0);
  float* outlb = (float*)(ws + OFF_OUTLB);
  float* outub = (float*)(ws + OFF_OUTUB);
  float* outf  = (float*)(ws + OFF_OUTF);

  // allow >64KB dynamic LDS (CDNA5 WGP: up to 320KB per workgroup)
  {
    auto* f1 = fused_mlp<1>; auto* f2 = fused_mlp<2>; auto* f4 = fused_mlp<4>;
    (void)hipFuncSetAttribute((const void*)f1,
        hipFuncAttributeMaxDynamicSharedMemorySize, (int)smem_bytes(1));
    (void)hipFuncSetAttribute((const void*)f2,
        hipFuncAttributeMaxDynamicSharedMemorySize, (int)smem_bytes(2));
    (void)hipFuncSetAttribute((const void*)f4,
        hipFuncAttributeMaxDynamicSharedMemorySize, (int)smem_bytes(4));
  }

  const int prepN = 3 * KP * KP;
  prep_weights<<<(prepN + 255) / 256, 256, 0, stream>>>(W2, W3, W4, wt);

  fused_mlp<1><<<N0_CNT / ROWS, 256, smem_bytes(1), stream>>>(
      X0, wt, W1, b1, b2, b3, b4, W5, b5, out0);
  fused_mlp<2><<<NB_CNT / ROWS, 256, smem_bytes(2), stream>>>(
      X_lb, wt, W1, b1, b2, b3, b4, W5, b5, outlb);
  fused_mlp<2><<<NB_CNT / ROWS, 256, smem_bytes(2), stream>>>(
      X_ub, wt, W1, b1, b2, b3, b4, W5, b5, outub);
  fused_mlp<4><<<NF_CNT / ROWS, 256, smem_bytes(4), stream>>>(
      X_f, wt, W1, b1, b2, b3, b4, W5, b5, outf);

  finalize_loss<<<1, 256, 0, stream>>>(out0, outlb, outub, outf, u0t, v0t,
                                       (float*)d_out);
}
